// NetHSP_GIN_16269336118021
// MI455X (gfx1250) — matrix operand order repaired and checked by reading, not compiled, not run on hardware
//
#include <hip/hip_runtime.h>

typedef __attribute__((ext_vector_type(16))) _Float16 v16h;
typedef __attribute__((ext_vector_type(8)))  _Float16 v8h;
typedef __attribute__((ext_vector_type(8)))  float    v8f;

#define NN 50000
#define EE 800000
#define DD 5
#define CC 16
#define GG 256
#define LL 2
#define HH 64
#define BN_EPS 1e-5f

// ---------------------------------------------------------------- utilities
__global__ void k_fill_u32(unsigned* __restrict__ p, unsigned v, int n) {
  int i = blockIdx.x * blockDim.x + threadIdx.x;
  if (i < n) p[i] = v;
}

__global__ void k_cvt_f16(const float* __restrict__ s, _Float16* __restrict__ d, int n) {
  int i = blockIdx.x * blockDim.x + threadIdx.x;
  if (i < n) d[i] = (_Float16)s[i];
}

__global__ void k_softmax_hop(const float* __restrict__ hc, float* __restrict__ w) {
  int l = threadIdx.x;
  if (l < LL) {
    const float* c = hc + l * DD;
    float m = c[0];
    for (int d = 1; d < DD; ++d) m = fmaxf(m, c[d]);
    float e[DD], s = 0.f;
    for (int d = 0; d < DD; ++d) { e[d] = __expf(c[d] - m); s += e[d]; }
    for (int d = 0; d < DD; ++d) w[l * 8 + d] = e[d] / s;
  }
}

// ------------------------------------------------------------- WMMA GEMM
// C[M,NCOL] = A[M,64] (f16) x B[64,NCOL] (f16) (+bias), f32 accumulate.
// One wave per 16-row tile; B pre-swizzled into per-lane WMMA fragment order
// in LDS so each fragment is one contiguous 32B ds read.
template <int NTILES>
__global__ void __launch_bounds__(256)
k_gemm(const _Float16* __restrict__ A, const _Float16* __restrict__ B,
       const float* __restrict__ bias, float* __restrict__ C, int Mtiles) {
  const int NCOL = NTILES * 16;
  __shared__ __align__(32) _Float16 bfrag[2 * NTILES * 32 * 16];

  // cooperative fill: bfrag[((kkidx*NTILES+nt)*32+lane)*16 + j]
  //   lanes 0-15: N=lane,    K = kk + (j<8 ? j : 16+j-8)      (j = 0..15)
  //   lanes16-31: N=lane-16, K = kk + (j<8 ? 8+j : 24+j-8)
  for (int i = threadIdx.x; i < 2 * NTILES * 512; i += blockDim.x) {
    int j     = i & 15;
    int lane  = (i >> 4) & 31;
    int t     = i >> 9;
    int nt    = t % NTILES;
    int kkidx = t / NTILES;
    int krow  = kkidx * 32 + ((j >> 3) << 4) + ((lane >> 4) << 3) + (j & 7);
    int ncol  = nt * 16 + (lane & 15);
    bfrag[i] = B[krow * NCOL + ncol];
  }
  __syncthreads();

  int lane = threadIdx.x & 31;
  int tile = blockIdx.x * (blockDim.x >> 5) + (threadIdx.x >> 5);
  if (tile >= Mtiles) return;          // wave-uniform; EXEC stays all-ones

  int half = lane >> 4;
  int nloc = lane & 15;
  int row0 = tile << 4;

  v8f acc[NTILES];
  for (int nt = 0; nt < NTILES; ++nt) {
    float bv = bias ? bias[nt * 16 + nloc] : 0.f;
    #pragma unroll
    for (int r = 0; r < 8; ++r) acc[nt][r] = bv;
  }

  const _Float16* Arow = A + (size_t)(row0 + nloc) * 64;
  #pragma unroll
  for (int kkidx = 0; kkidx < 2; ++kkidx) {
    int kk = kkidx * 32;
    // A 16x32 f16 layout: VGPR0-3 = K(8h..8h+7), VGPR4-7 = K(16+8h..16+8h+7)
    v8h alo = *(const v8h*)(Arow + kk + 8 * half);
    v8h ahi = *(const v8h*)(Arow + kk + 16 + 8 * half);
    v16h a;
    #pragma unroll
    for (int q = 0; q < 8; ++q) { a[q] = alo[q]; a[8 + q] = ahi[q]; }
    #pragma unroll
    for (int nt = 0; nt < NTILES; ++nt) {
      v16h b = *(const v16h*)(&bfrag[((kkidx * NTILES + nt) * 32 + lane) * 16]);
      acc[nt] = __builtin_amdgcn_wmma_f32_16x16x32_f16(
          false, a, false, b, (short)0, acc[nt], false, false);
    }
  }

  // D layout: VGPR r holds (M = half*8 + r, N = nloc)
  int mo = row0 + half * 8;
  for (int nt = 0; nt < NTILES; ++nt) {
    int n = nt * 16 + nloc;
    #pragma unroll
    for (int r = 0; r < 8; ++r) C[(size_t)(mo + r) * NCOL + n] = acc[nt][r];
  }
}

// ----------------------------------------------- batchnorm statistics / apply
__global__ void k_colstats(const float* __restrict__ Y, float* __restrict__ stats, int n) {
  __shared__ float ss[HH], ss2[HH];
  if (threadIdx.x < HH) { ss[threadIdx.x] = 0.f; ss2[threadIdx.x] = 0.f; }
  __syncthreads();
  int base = blockIdx.x * blockDim.x + threadIdx.x;
  int stride = gridDim.x * blockDim.x;   // must be multiple of 64
  float s = 0.f, s2 = 0.f;
  for (int i = base; i < n; i += stride) { float v = Y[i]; s += v; s2 += v * v; }
  atomicAdd(&ss[base & 63], s);
  atomicAdd(&ss2[base & 63], s2);
  __syncthreads();
  if (threadIdx.x < HH) {
    atomicAdd(&stats[threadIdx.x], ss[threadIdx.x]);
    atomicAdd(&stats[HH + threadIdx.x], ss2[threadIdx.x]);
  }
}

__global__ void k_bn_relu(const float* __restrict__ Y, const float* __restrict__ stats,
                          const float* __restrict__ g, const float* __restrict__ be,
                          float* __restrict__ of32, _Float16* __restrict__ of16, int n) {
  int i = blockIdx.x * blockDim.x + threadIdx.x;
  if (i >= n) return;
  int c = i & 63;
  const float invN = 1.f / (float)NN;
  float mu  = stats[c] * invN;
  float var = stats[HH + c] * invN - mu * mu;
  float v = (Y[i] - mu) * rsqrtf(var + BN_EPS) * g[c] + be[c];
  v = fmaxf(v, 0.f);
  if (of32) of32[i] = v;
  of16[i] = (_Float16)v;
}

// --------------------------------------------------------- edge propagation
// comb[dst] += softmax(hop_coef)[hop] * h[src]   (weights distribute over the
// segment sum, so the [N,D,H] agg tensor is never materialized)
__global__ void k_edge(const float* __restrict__ h, const int* __restrict__ src,
                       const int* __restrict__ dst, const int* __restrict__ hop,
                       const float* __restrict__ w, float* __restrict__ comb) {
  int t = blockIdx.x * blockDim.x + threadIdx.x;
  if (t >= EE * 16) return;
  int e = t >> 4;
  int c = (t & 15) * 4;
  float wv = w[hop[e]];
  const float4 v = *(const float4*)(h + (size_t)src[e] * HH + c);
  float* p = comb + (size_t)dst[e] * HH + c;
  atomicAdd(p + 0, v.x * wv);
  atomicAdd(p + 1, v.y * wv);
  atomicAdd(p + 2, v.z * wv);
  atomicAdd(p + 3, v.w * wv);
}

__global__ void k_add_cvt(const float* __restrict__ h, const float* __restrict__ comb,
                          _Float16* __restrict__ d, int n) {
  int i = blockIdx.x * blockDim.x + threadIdx.x;
  if (i < n) d[i] = (_Float16)(h[i] + comb[i]);
}

// ------------------------------------------------------------- max pooling
__device__ __forceinline__ unsigned fkey(float x) {
  unsigned u = __float_as_uint(x);
  return (u & 0x80000000u) ? ~u : (u | 0x80000000u);
}
__device__ __forceinline__ float unfkey(unsigned k) {
  unsigned u = (k & 0x80000000u) ? (k & 0x7FFFFFFFu) : ~k;
  return __uint_as_float(u);
}

__global__ void k_count(const int* __restrict__ batch, int* __restrict__ cnt) {
  int i = blockIdx.x * blockDim.x + threadIdx.x;
  if (i < NN) atomicAdd(&cnt[batch[i]], 1);
}

__global__ void k_pool_max(const float* __restrict__ v, const int* __restrict__ batch,
                           unsigned* __restrict__ keys, int shift, int n) {
  int i = blockIdx.x * blockDim.x + threadIdx.x;
  if (i >= n) return;
  int node = i >> shift;
  int c = i & ((1 << shift) - 1);
  atomicMax(&keys[(batch[node] << shift) + c], fkey(v[i]));
}

__global__ void k_finalize(const unsigned* __restrict__ keys, const int* __restrict__ cnt,
                           float* __restrict__ out, int shift, int n) {
  int i = blockIdx.x * blockDim.x + threadIdx.x;
  if (i >= n) return;
  out[i] = (cnt[i >> shift] > 0) ? unfkey(keys[i]) : 0.f;
}

__global__ void k_readout_add(const float* __restrict__ pooled, const float* __restrict__ W,
                              const float* __restrict__ b, float* __restrict__ out) {
  int i = blockIdx.x * blockDim.x + threadIdx.x;
  if (i >= GG * CC) return;
  int g = i >> 4, c = i & 15;
  float s = b[c];
  const float* pr = pooled + g * HH;
  #pragma unroll 8
  for (int k = 0; k < HH; ++k) s += pr[k] * W[k * CC + c];
  out[i] += s;
}

// ----------------------------------------------------------------- driver
#define GRID(n) (((n) + 255) / 256)

extern "C" void kernel_launch(void* const* d_in, const int* in_sizes, int n_in,
                              void* d_out, int out_size, void* d_ws, size_t ws_size,
                              hipStream_t stream) {
  (void)in_sizes; (void)n_in; (void)out_size; (void)ws_size;
  const float* x        = (const float*)d_in[0];
  const int*   ei       = (const int*)d_in[1];
  const int*   src      = ei;
  const int*   dst      = ei + EE;
  const int*   hop      = (const int*)d_in[2];
  const int*   batch    = (const int*)d_in[3];
  const float* mlp_W1   = (const float*)d_in[4];
  const float* mlp_g1   = (const float*)d_in[6];
  const float* mlp_be1  = (const float*)d_in[7];
  const float* mlp_W2   = (const float*)d_in[8];
  const float* mlp_g2   = (const float*)d_in[10];
  const float* mlp_be2  = (const float*)d_in[11];
  const float* init_W   = (const float*)d_in[12];
  const float* init_b   = (const float*)d_in[13];
  const float* hop_coef = (const float*)d_in[14];
  const float* gin_W1   = (const float*)d_in[15];
  const float* gin_g1   = (const float*)d_in[17];
  const float* gin_be1  = (const float*)d_in[18];
  const float* gin_W2   = (const float*)d_in[19];
  const float* gin_g2   = (const float*)d_in[21];
  const float* gin_be2  = (const float*)d_in[22];
  const float* lin_W    = (const float*)d_in[23];
  const float* lin_b    = (const float*)d_in[24];
  float* out = (float*)d_out;

  char* wsb = (char*)d_ws;
  size_t off = 0;
  auto alloc = [&](size_t bytes) -> void* {
    void* p = wsb + off;
    off += (bytes + 255) & ~(size_t)255;
    return p;
  };
  _Float16* fA    = (_Float16*)alloc((size_t)NN * HH * 2);  // GEMM A input (f16)
  float*    Y     = (float*)alloc((size_t)NN * HH * 4);     // pre-BN linear output
  float*    hbuf  = (float*)alloc((size_t)NN * HH * 4);     // master h (f32)
  float*    comb  = (float*)alloc((size_t)NN * HH * 4);     // weighted message sum
  float*    Z     = (float*)alloc((size_t)NN * CC * 4);     // init readout pre-pool
  float*    stats = (float*)alloc(2 * HH * 4);
  float*    wbuf  = (float*)alloc(16 * 4);                  // softmax(hop_coef), 8/layer
  int*      cnt   = (int*)alloc(GG * 4);
  unsigned* keys  = (unsigned*)alloc((size_t)GG * HH * 4);
  float*    pooled= (float*)alloc((size_t)GG * HH * 4);
  _Float16* w1h   = (_Float16*)alloc((size_t)HH * HH * 2);
  _Float16* w2h   = (_Float16*)alloc((size_t)HH * HH * 2);
  _Float16* gw1h  = (_Float16*)alloc((size_t)LL * HH * HH * 2);
  _Float16* gw2h  = (_Float16*)alloc((size_t)LL * HH * HH * 2);
  _Float16* iwh   = (_Float16*)alloc((size_t)HH * CC * 2);

  const int Mtiles   = NN / 16;          // 3125 (exact)
  const int gemmGrid = (Mtiles + 7) / 8; // 8 waves / block

  // weight + input conversions
  k_cvt_f16<<<GRID(NN * HH), 256, 0, stream>>>(x, fA, NN * HH);
  k_cvt_f16<<<GRID(HH * HH), 256, 0, stream>>>(mlp_W1, w1h, HH * HH);
  k_cvt_f16<<<GRID(HH * HH), 256, 0, stream>>>(mlp_W2, w2h, HH * HH);
  k_cvt_f16<<<GRID(LL * HH * HH), 256, 0, stream>>>(gin_W1, gw1h, LL * HH * HH);
  k_cvt_f16<<<GRID(LL * HH * HH), 256, 0, stream>>>(gin_W2, gw2h, LL * HH * HH);
  k_cvt_f16<<<GRID(HH * CC), 256, 0, stream>>>(init_W, iwh, HH * CC);
  k_softmax_hop<<<1, 32, 0, stream>>>(hop_coef, wbuf);
  k_fill_u32<<<1, 256, 0, stream>>>((unsigned*)cnt, 0u, GG);
  k_count<<<GRID(NN), 256, 0, stream>>>(batch, cnt);

  // ---- initial MLP (linear biases absorbed by BN) ----
  k_gemm<4><<<gemmGrid, 256, 0, stream>>>(fA, w1h, nullptr, Y, Mtiles);
  k_fill_u32<<<1, 128, 0, stream>>>((unsigned*)stats, 0u, 2 * HH);
  k_colstats<<<512, 256, 0, stream>>>(Y, stats, NN * HH);
  k_bn_relu<<<GRID(NN * HH), 256, 0, stream>>>(Y, stats, mlp_g1, mlp_be1, nullptr, fA, NN * HH);
  k_gemm<4><<<gemmGrid, 256, 0, stream>>>(fA, w2h, nullptr, Y, Mtiles);
  k_fill_u32<<<1, 128, 0, stream>>>((unsigned*)stats, 0u, 2 * HH);
  k_colstats<<<512, 256, 0, stream>>>(Y, stats, NN * HH);
  k_bn_relu<<<GRID(NN * HH), 256, 0, stream>>>(Y, stats, mlp_g2, mlp_be2, hbuf, fA, NN * HH);

  // ---- init readout: out = pool_max(h @ init_W + init_b) ----
  k_gemm<1><<<gemmGrid, 256, 0, stream>>>(fA, iwh, init_b, Z, Mtiles);
  k_fill_u32<<<GRID(GG * CC), 256, 0, stream>>>(keys, 0x007FFFFFu /* key(-inf) */, GG * CC);
  k_pool_max<<<GRID(NN * CC), 256, 0, stream>>>(Z, batch, keys, 4, NN * CC);
  k_finalize<<<GRID(GG * CC), 256, 0, stream>>>(keys, cnt, out, 4, GG * CC);

  // ---- GIN-HSP layers ----
  for (int l = 0; l < LL; ++l) {
    k_fill_u32<<<GRID(NN * HH), 256, 0, stream>>>((unsigned*)comb, 0u, NN * HH);
    k_edge<<<GRID(EE * 16), 256, 0, stream>>>(hbuf, src, dst, hop, wbuf + l * 8, comb);
    k_add_cvt<<<GRID(NN * HH), 256, 0, stream>>>(hbuf, comb, fA, NN * HH);

    k_gemm<4><<<gemmGrid, 256, 0, stream>>>(fA, gw1h + l * HH * HH, nullptr, Y, Mtiles);
    k_fill_u32<<<1, 128, 0, stream>>>((unsigned*)stats, 0u, 2 * HH);
    k_colstats<<<512, 256, 0, stream>>>(Y, stats, NN * HH);
    k_bn_relu<<<GRID(NN * HH), 256, 0, stream>>>(Y, stats, gin_g1 + l * HH, gin_be1 + l * HH,
                                                 nullptr, fA, NN * HH);

    k_gemm<4><<<gemmGrid, 256, 0, stream>>>(fA, gw2h + l * HH * HH, nullptr, Y, Mtiles);
    k_fill_u32<<<1, 128, 0, stream>>>((unsigned*)stats, 0u, 2 * HH);
    k_colstats<<<512, 256, 0, stream>>>(Y, stats, NN * HH);
    k_bn_relu<<<GRID(NN * HH), 256, 0, stream>>>(Y, stats, gin_g2 + l * HH, gin_be2 + l * HH,
                                                 hbuf, fA, NN * HH);

    k_fill_u32<<<GRID(GG * HH), 256, 0, stream>>>(keys, 0x007FFFFFu, GG * HH);
    k_pool_max<<<GRID(NN * HH), 256, 0, stream>>>(hbuf, batch, keys, 6, NN * HH);
    k_finalize<<<GRID(GG * HH), 256, 0, stream>>>(keys, cnt, pooled, 6, GG * HH);
    k_readout_add<<<GRID(GG * CC), 256, 0, stream>>>(pooled, lin_W + l * HH * CC,
                                                     lin_b + l * CC, out);
  }
}
